// Cross_attention_13417477833532
// MI455X (gfx1250) — matrix operand order repaired and checked by reading, not compiled, not run on hardware
//
#include <hip/hip_runtime.h>

typedef __attribute__((ext_vector_type(16))) __bf16        v16bf;
typedef __attribute__((ext_vector_type(8)))  float         v8f;
typedef __attribute__((ext_vector_type(4)))  unsigned int  u32x4;
typedef __attribute__((ext_vector_type(4)))  int           i32x4;
typedef __attribute__((ext_vector_type(4)))  float         f4;
typedef __attribute__((ext_vector_type(4)))  __bf16        bf4;

#define BATCH   8
#define SEQ     2048
#define MTOK    16
#define DIM     256
#define HEADS   8
#define DHEAD   64
#define INNER   512     // HEADS*DHEAD
#define KVCOLS  1024    // 2*INNER
#define SCALE   0.125f  // 64^-0.5
#define NPOS    4       // positions per workgroup (B-fragment reuse factor)

union FragU { v16bf v; u32x4 q[2]; };

// LDS layout (bytes). yF (f32 async staging, 65536B) aliases the start of the
// kvS region: yF is fully consumed (converted to yA) before kvS is written.
#define OFF_YA    0                          // [4][16][256] bf16 : 32768
#define OFF_XROW  32768                      // [4][256]     f32  : 4096
#define OFF_KVS   36864                      // [4][16][1024] bf16: 131072
#define OFF_YF    OFF_KVS                    // [4][16][256] f32  : 65536 (aliased)
#define OFF_QV    167936                     // [4][512]     f32  : 8192
#define OFF_DOTS  176128                     // [4][8][16]   f32  : 2048
#define OFF_ATTN  178176                     // [4][8][16]   f32  : 2048
#define OFF_OIN   180224                     // [4][512]     f32  : 8192
#define SMEM_BYTES 188416

#if __has_builtin(__builtin_amdgcn_global_load_async_to_lds_b128)
#define HAVE_ASYNC_LDS 1
#else
#define HAVE_ASYNC_LDS 0
#endif

// Builtin signature (from hipcc diagnostic): param 0 is AS(1) int4*, so cast
// both pointers to address-space-qualified int4 vector pointers.
#define ASYNC_CP_B128(gsrc, ldst)                                        \
    __builtin_amdgcn_global_load_async_to_lds_b128(                      \
        (__attribute__((address_space(1))) i32x4*)(gsrc),                \
        (__attribute__((address_space(3))) i32x4*)(ldst), 0, 0)

__device__ __forceinline__ void wait_async_zero() {
#if __has_builtin(__builtin_amdgcn_s_wait_asynccnt)
    __builtin_amdgcn_s_wait_asynccnt(0);
#else
    asm volatile("s_wait_asynccnt 0" ::: "memory");
#endif
}

// ---------------------------------------------------------------------------
// Kernel 0: convert W_kv [K=256][N=1024] f32 -> bf16, transposed to [N][K]
// so each WMMA B-fragment is one contiguous 32B per-lane run (2x b128).
// ---------------------------------------------------------------------------
__global__ __launch_bounds__(256)
void wkv_convert_kernel(const float* __restrict__ wkv, __bf16* __restrict__ wkvT) {
    int idx = blockIdx.x * 256 + threadIdx.x;   // 0 .. DIM*KVCOLS-1
    int k = idx >> 10;                          // 0..255
    int j = idx & (KVCOLS - 1);                 // 0..1023
    wkvT[j * DIM + k] = (__bf16)wkv[idx];
}

// ---------------------------------------------------------------------------
// Kernel 1: fused cross-attention, NPOS=4 (b,n) positions per workgroup.
// 256 threads = 8 wave32. Async y->LDS copy overlapped with q GEMV, then
// kv GEMM [4x16 x 256]@[256 x 1024] via WMMA bf16 with 4x B-fragment reuse.
// ---------------------------------------------------------------------------
__global__ __launch_bounds__(256)
void cross_attn_kernel(const float*  __restrict__ x,
                       const float*  __restrict__ y,
                       const float*  __restrict__ W_q,
                       const __bf16* __restrict__ wkvT,
                       const float*  __restrict__ W_out,
                       const float*  __restrict__ b_out,
                       float*        __restrict__ out) {
    extern __shared__ __align__(16) unsigned char smem_raw[];
    __bf16* yA    = (__bf16*)(smem_raw + OFF_YA);
    float*  xrow  = (float*)(smem_raw + OFF_XROW);
    __bf16* kvS   = (__bf16*)(smem_raw + OFF_KVS);
    float*  yF    = (float*)(smem_raw + OFF_YF);
    float*  qv    = (float*)(smem_raw + OFF_QV);
    float*  dots  = (float*)(smem_raw + OFF_DOTS);
    float*  attnW = (float*)(smem_raw + OFF_ATTN);
    float*  oin   = (float*)(smem_raw + OFF_OIN);

    const int pos0 = blockIdx.x * NPOS;  // first of 4 consecutive (b,n) positions
    const int t    = threadIdx.x;        // 0..255
    const int l    = t & 31;             // lane
    const int w    = t >> 5;             // wave 0..7
    const int lcol = l & 15;             // N column within 16-wide tile / A row
    const int hi   = l >> 4;             // lane-half select (0 or 1)

    const float* ybase = y + (size_t)pos0 * (MTOK * DIM);   // 4 tiles contiguous

    // ---- (1) launch async HBM -> LDS copy of raw f32 y tiles (64 KB) ----
#if HAVE_ASYNC_LDS
    {
        const f4* ysrc = (const f4*)ybase;
        f4*       ydst = (f4*)yF;
#pragma unroll
        for (int i = 0; i < (NPOS * MTOK * DIM) / (4 * 256); ++i) {   // 16 iters
            int vi = t + i * 256;
            ASYNC_CP_B128(ysrc + vi, ydst + vi);
        }
    }
#endif

    // ---- (2) stage x rows ----
#pragma unroll
    for (int i = 0; i < NPOS; ++i)
        xrow[t + i * 256] = x[(size_t)pos0 * DIM + t + i * 256];
    __syncthreads();   // xrow visible; async y copy still in flight

    // ---- (3) q = x @ W_q : 4x256 @ 256x512 GEMV (hides async y latency) ----
    {
        float s[NPOS][2];
#pragma unroll
        for (int p = 0; p < NPOS; ++p) { s[p][0] = 0.f; s[p][1] = 0.f; }
#pragma unroll 4
        for (int k = 0; k < DIM; ++k) {
            float w0 = W_q[(size_t)k * INNER + t];
            float w1 = W_q[(size_t)k * INNER + t + 256];
#pragma unroll
            for (int p = 0; p < NPOS; ++p) {
                float xv = xrow[p * DIM + k];
                s[p][0] += xv * w0;
                s[p][1] += xv * w1;
            }
        }
#pragma unroll
        for (int p = 0; p < NPOS; ++p) {
            qv[p * INNER + t]       = s[p][0];
            qv[p * INNER + t + 256] = s[p][1];
        }
    }

    // ---- (4) complete async copy, convert y f32 -> bf16 WMMA A layout ----
#if HAVE_ASYNC_LDS
    wait_async_zero();
    __syncthreads();   // all waves' async data in LDS
    {
        const f4* src = (const f4*)yF;
        bf4*      dst = (bf4*)yA;
#pragma unroll
        for (int i = 0; i < (NPOS * MTOK * DIM) / (4 * 256); ++i) {   // 16 iters
            int vi = t + i * 256;
            f4 v = src[vi];
            bf4 o;
            o[0] = (__bf16)v[0]; o[1] = (__bf16)v[1];
            o[2] = (__bf16)v[2]; o[3] = (__bf16)v[3];
            dst[vi] = o;
        }
    }
#else
#pragma unroll
    for (int i = 0; i < (NPOS * MTOK * DIM) / 256; ++i) {             // 64 iters
        int e = t + i * 256;
        yA[e] = (__bf16)ybase[e];
    }
#endif
    __syncthreads();   // yA ready; yF region now dead -> kvS may overwrite

    // ---- (5) kv GEMM: 4 x [16x256] @ [256x1024] -> kvS (bf16), WMMA ----
    auto loadB = [&](int half, int kk, int tt, FragU& f) {
        int col = half * 512 + w * 64 + tt * 16 + lcol;
        const u32x4* bp = (const u32x4*)(wkvT + (size_t)col * DIM + kk * 32 + hi * 16);
        f.q[0] = bp[0];
        f.q[1] = bp[1];
    };
    auto loadA = [&](int p, int kk, FragU& f) {
        const u32x4* ap = (const u32x4*)(yA + (p * MTOK + lcol) * DIM + kk * 32 + hi * 16);
        f.q[0] = ap[0];     // klocal base .. base+7
        f.q[1] = ap[1];     // klocal base+8 .. base+15
    };

#pragma unroll
    for (int half = 0; half < 2; ++half) {
        v8f acc[NPOS][4];
#pragma unroll
        for (int p = 0; p < NPOS; ++p)
#pragma unroll
            for (int tt = 0; tt < 4; ++tt) acc[p][tt] = (v8f)0.0f;

        FragU fa[NPOS];
        FragU fb[2];
        loadB(half, 0, 0, fb[0]);           // prime the B pipeline
        int buf = 0;
#pragma unroll
        for (int kk = 0; kk < 8; ++kk) {    // K chunks of 32
#pragma unroll
            for (int p = 0; p < NPOS; ++p) loadA(p, kk, fa[p]);
#pragma unroll
            for (int tt = 0; tt < 4; ++tt) {
                int it = kk * 4 + tt;
                if (it < 31)                // prefetch next B fragment
                    loadB(half, (it + 1) >> 2, (it + 1) & 3, fb[buf ^ 1]);
#pragma unroll
                for (int p = 0; p < NPOS; ++p)
                    acc[p][tt] = __builtin_amdgcn_wmma_f32_16x16x32_bf16(
                        false, fa[p].v, false, fb[buf].v, (short)0,
                        acc[p][tt], false, false);
                buf ^= 1;
            }
        }
        // store: C/D layout M = r + 8*hi, N = lane&15
#pragma unroll
        for (int p = 0; p < NPOS; ++p)
#pragma unroll
            for (int tt = 0; tt < 4; ++tt) {
                int col = half * 512 + w * 64 + tt * 16 + lcol;
#pragma unroll
                for (int r = 0; r < 8; ++r)
                    kvS[(size_t)(p * MTOK + r + hi * 8) * KVCOLS + col] =
                        (__bf16)acc[p][tt][r];
            }
    }
    __syncthreads();

    // ---- (6) dots[p][h][m] = SCALE * q[p,h,:] . k[p,m,h,:] ----
#pragma unroll
    for (int ii = 0; ii < 2; ++ii) {
        int idx = t + ii * 256;              // 0..511 = p*128 + h*16 + m
        int p = idx >> 7, h = (idx >> 4) & 7, mm = idx & 15;
        float s = 0.f;
#pragma unroll 8
        for (int d = 0; d < DHEAD; ++d)
            s += qv[p * INNER + h * DHEAD + d] *
                 (float)kvS[(size_t)(p * MTOK + mm) * KVCOLS + h * DHEAD + d];
        dots[(p * HEADS + h) * MTOK + mm] = s * SCALE;
    }
    __syncthreads();

    // ---- (7) softmax over m per (p, head) ----
    if (t < NPOS * HEADS) {
        float mx = -1e30f;
#pragma unroll
        for (int mm = 0; mm < MTOK; ++mm) mx = fmaxf(mx, dots[t * MTOK + mm]);
        float e[MTOK], s = 0.f;
#pragma unroll
        for (int mm = 0; mm < MTOK; ++mm) { e[mm] = __expf(dots[t * MTOK + mm] - mx); s += e[mm]; }
        float inv = 1.f / s;
#pragma unroll
        for (int mm = 0; mm < MTOK; ++mm) attnW[t * MTOK + mm] = e[mm] * inv;
    }
    __syncthreads();

    // ---- (8) out_inner[p][h*64+d] = sum_m attn[p][h][m] * v[p,m,h,d] ----
#pragma unroll
    for (int ii = 0; ii < (NPOS * INNER) / 256; ++ii) {     // 8 iters
        int idx = t + ii * 256;              // p*512 + j
        int p = idx >> 9, j = idx & 511;
        int h = j >> 6, d = j & 63;
        float s = 0.f;
#pragma unroll
        for (int mm = 0; mm < MTOK; ++mm)
            s += attnW[(p * HEADS + h) * MTOK + mm] *
                 (float)kvS[(size_t)(p * MTOK + mm) * KVCOLS + INNER + h * DHEAD + d];
        oin[idx] = s;
    }
    __syncthreads();

    // ---- (9) final: out[p][c] = b_out[c] + oin[p,:] . W_out[:,c] ----
    {
        float s[NPOS];
        float bo = b_out[t];
#pragma unroll
        for (int p = 0; p < NPOS; ++p) s[p] = bo;
#pragma unroll 4
        for (int j = 0; j < INNER; ++j) {
            float wv = W_out[(size_t)j * DIM + t];
#pragma unroll
            for (int p = 0; p < NPOS; ++p)
                s[p] += oin[p * INNER + j] * wv;
        }
#pragma unroll
        for (int p = 0; p < NPOS; ++p)
            out[(size_t)(pos0 + p) * DIM + t] = s[p];
    }
}

// ---------------------------------------------------------------------------
extern "C" void kernel_launch(void* const* d_in, const int* in_sizes, int n_in,
                              void* d_out, int out_size, void* d_ws, size_t ws_size,
                              hipStream_t stream) {
    const float* x     = (const float*)d_in[0];
    const float* y     = (const float*)d_in[1];
    const float* W_q   = (const float*)d_in[2];
    const float* W_kv  = (const float*)d_in[3];
    const float* W_out = (const float*)d_in[4];
    const float* b_out = (const float*)d_in[5];
    float* out = (float*)d_out;

    __bf16* wkvT = (__bf16*)d_ws;   // 1024*256 bf16 = 512 KB

    wkv_convert_kernel<<<(DIM * KVCOLS) / 256, 256, 0, stream>>>(W_kv, wkvT);

    cross_attn_kernel<<<(BATCH * SEQ) / NPOS, 256, SMEM_BYTES, stream>>>(
        x, y, W_q, wkvT, W_out, b_out, out);
}
